// ITERMEM_47270410060249
// MI455X (gfx1250) — compile-verified, matrix operand order checked by reading, not run on hardware
//
#include <hip/hip_runtime.h>
#include <math.h>
#include <stdint.h>

typedef __attribute__((ext_vector_type(16))) _Float16 v16h;
typedef __attribute__((ext_vector_type(8)))  _Float16 v8h;
typedef __attribute__((ext_vector_type(8)))  float    v8f;
typedef __attribute__((ext_vector_type(4)))  uint32_t v4u;
typedef __attribute__((ext_vector_type(8)))  uint32_t v8u;

union V16H { v16h v; v8h h[2]; };

#define G_   128
#define S_   32
#define P_   16
#define D_   768
#define C_   256
#define NH_  4
#define OUT_ 7
#define GP   (G_ * P_)   // 2048

// ---------------------------------------------------------------------------
// TDM: issue a 2D tensor_load_to_lds of a [rows x 64] f16 tile.
//   gptr: tile start (row-major, rowStride halfs between rows)
//   lds : destination (tile stored contiguously, [rows][64] halfs)
// D# per CDNA5 ISA (groups 0/1; 2D => VADDR2/3 omitted).
// All operands wave-uniform -> SGPR tuples via "s" constraints.
// ---------------------------------------------------------------------------
__device__ __forceinline__ void tdm_load_f16(const _Float16* gptr,
                                             int rowStrideElems,
                                             void* ldsPtr, uint32_t rows)
{
    uint64_t ga  = (uint64_t)(uintptr_t)gptr;          // byte address
    uint32_t lds = (uint32_t)(uintptr_t)ldsPtr;        // LDS byte offset
    // group 0: count=1 | lds_addr | global_addr | type=2
    v4u g0 = { 1u,
               lds,
               (uint32_t)ga,
               (uint32_t)(ga >> 32) | (2u << 30) };
    // group 1: data_size=2B(code1); tensor_dim0=64, tensor_dim1=rows (== tile);
    //          tile_dim0=64, tile_dim1=rows, tile_dim2=0; dim0_stride=rowStrideElems
    const uint32_t tc = 64u;
    v8u g1 = { (1u << 16),
               (tc & 0xffffu) << 16,
               ((tc >> 16) & 0xffffu) | ((rows & 0xffffu) << 16),
               ((rows >> 16) & 0xffffu) | (tc << 16),
               (rows & 0xffffu),
               (uint32_t)rowStrideElems,
               0u, 0u };
    asm volatile("tensor_load_to_lds %0, %1" : : "s"(g0), "s"(g1) : "memory");
}

// ---------------------------------------------------------------------------
// f16 WMMA GEMM, TDM-fed, double-buffered LDS.
//   A: f16 row-major [M,K], row stride lda (halfs)
//   B: f16 row-major [N,K], row stride ldb (halfs)   (i.e. C = A @ B^T)
//   EPI: 0 none, 1 +bias[n], 2 relu(x+bias[n]);  C16: store f16 else f32
// Block: 256 threads = 8 wave32. Tile 64(M) x 128(N), K-chunk 64.
// Wave w: rows (w&3)*16, cols (w>>2)*64 -> 2 K-steps x 4 N-frags = 8 WMMA/stage.
// Wave0 streams A tiles, wave1 streams B tiles via tensor_load_to_lds;
// steady state waits s_wait_tensorcnt 1 so the next DMA overlaps compute.
// ---------------------------------------------------------------------------
template<int EPI, int C16>
__global__ void gemm_tdm(const _Float16* __restrict__ A, int lda,
                         const _Float16* __restrict__ B, int ldb,
                         void* __restrict__ Cout, int ldc,
                         int K, const float* __restrict__ bias)
{
    __shared__ __align__(16) _Float16 Ah[2][64][64];    // 16 KB
    __shared__ __align__(16) _Float16 Bh[2][128][64];   // 32 KB

    const int tid  = threadIdx.x;
    const int lane = tid & 31;
    const int wave = tid >> 5;
    const int n0 = blockIdx.x * 128;
    const int m0 = blockIdx.y * 64;
    const int mw = (wave & 3) * 16;     // wave's row offset within tile
    const int nw = (wave >> 2) * 64;    // wave's col offset within tile

    v8f acc[4];
    for (int i = 0; i < 4; ++i) acc[i] = 0.0f;

    const int mrow = mw + (lane & 15);
    const int klo  = ((lane >> 4) & 1) * 8;   // lanes 0-15: K 0..7/16..23; 16-31: +8

    const int nchunks = K >> 6;

    // prologue: chunk 0 -> buffer 0
    if (wave == 0)      tdm_load_f16(A + (size_t)m0 * lda, lda, &Ah[0][0][0], 64u);
    else if (wave == 1) tdm_load_f16(B + (size_t)n0 * ldb, ldb, &Bh[0][0][0], 128u);

    for (int i = 0; i < nchunks; ++i) {
        const int cur = i & 1, nxt = cur ^ 1;
        // all waves done reading buf[nxt] (from iteration i-1) before DMA reuses it
        __syncthreads();
        if (i + 1 < nchunks) {
            const int k0 = (i + 1) << 6;
            if (wave == 0)
                tdm_load_f16(A + (size_t)m0 * lda + k0, lda, &Ah[nxt][0][0], 64u);
            else if (wave == 1)
                tdm_load_f16(B + (size_t)n0 * ldb + k0, ldb, &Bh[nxt][0][0], 128u);
            if (wave < 2) __builtin_amdgcn_s_wait_tensorcnt(1);  // chunk i landed
        } else {
            if (wave < 2) __builtin_amdgcn_s_wait_tensorcnt(0);
        }
        __syncthreads();    // publish buf[cur]

        for (int kk = 0; kk < 2; ++kk) {
            const int kb = kk * 32;
            V16H a;
            a.h[0] = *reinterpret_cast<const v8h*>(&Ah[cur][mrow][kb + klo]);
            a.h[1] = *reinterpret_cast<const v8h*>(&Ah[cur][mrow][kb + 16 + klo]);
            for (int nb = 0; nb < 4; ++nb) {
                const int nr = nw + nb * 16 + (lane & 15);
                V16H b;
                b.h[0] = *reinterpret_cast<const v8h*>(&Bh[cur][nr][kb + klo]);
                b.h[1] = *reinterpret_cast<const v8h*>(&Bh[cur][nr][kb + 16 + klo]);
                acc[nb] = __builtin_amdgcn_wmma_f32_16x16x32_f16(
                    false, a.v, false, b.v, (short)0, acc[nb], false, false);
            }
        }
    }

    // store: VGPR v -> row m0+mw+v (+8 for lanes 16-31), col nw+nb*16+(lane&15)
    const int rbase = m0 + mw + ((lane >> 4) & 1) * 8;
    const int ncol  = lane & 15;
    for (int nb = 0; nb < 4; ++nb) {
        int n = n0 + nw + nb * 16 + ncol;
        float bv = (EPI > 0) ? bias[n] : 0.0f;
        for (int v = 0; v < 8; ++v) {
            float val = acc[nb][v] + bv;
            if (EPI == 2) val = fmaxf(val, 0.0f);
            if (C16) ((_Float16*)Cout)[(size_t)(rbase + v) * ldc + n] = (_Float16)val;
            else     ((float*)Cout)[(size_t)(rbase + v) * ldc + n] = val;
        }
    }
}

// ---------------------------------------------------------------------------
// Prep: weights -> f16 [N][K] layout (once per launch)
// ---------------------------------------------------------------------------
__global__ void k_fill0(float* p, int n) {
    int i = blockIdx.x * 256 + threadIdx.x;
    if (i < n) p[i] = 0.0f;
}
__global__ void k_fillh0(_Float16* p, int n) {
    int i = blockIdx.x * 256 + threadIdx.x;
    if (i < n) p[i] = (_Float16)0.0f;
}
__global__ void k_f32_to_f16(const float* __restrict__ s, _Float16* __restrict__ d, int n) {
    int i = blockIdx.x * 256 + threadIdx.x;
    if (i < n) d[i] = (_Float16)s[i];
}
// gat_W [K=768][N=1024] -> gatWh [1024][768] f16 (transpose)
__global__ void k_prep_gatW(const float* __restrict__ W, _Float16* __restrict__ Wh) {
    int i = blockIdx.x * 256 + threadIdx.x;     // i = n*768 + k
    if (i < (NH_ * C_) * D_) {
        int n = i / D_, k = i % D_;
        Wh[i] = (_Float16)W[(size_t)k * (NH_ * C_) + n];
    }
}
// sp_Wsum[o][c] = sp_W[o][c] + sp_W[o][D+c]   (concat [pooled0,pooled0] folded)
__global__ void k_spwsum_h(const float* __restrict__ spW, _Float16* __restrict__ spWs) {
    int i = blockIdx.x * 256 + threadIdx.x;
    if (i < D_ * D_) {
        int o = i / D_, c = i % D_;
        spWs[i] = (_Float16)(spW[(size_t)o * (2 * D_) + c] +
                             spW[(size_t)o * (2 * D_) + D_ + c]);
    }
}
// Xbufh[g*16+p][d] = x[g][0][p][d]  (only x[:,0] is live in the reference)
__global__ void k_gather_x0(const float* __restrict__ x, _Float16* __restrict__ Xh) {
    int i = blockIdx.x * 256 + threadIdx.x;
    if (i < GP * D_) {
        int r = i / D_, d = i % D_;
        int g = r >> 4, p = r & 15;
        Xh[i] = (_Float16)x[(((size_t)g * S_ + 0) * P_ + p) * D_ + d];
    }
}
__global__ void k_scatter_spk(const _Float16* __restrict__ spk, _Float16* __restrict__ Xh) {
    int i = blockIdx.x * 256 + threadIdx.x;
    if (i < G_ * D_) {
        int g = i / D_, d = i % D_;
        Xh[(size_t)(g * P_) * D_ + d] = spk[i];
    }
}

// ---------------------------------------------------------------------------
// GAT pointwise: scores + star-graph softmax combine
// ---------------------------------------------------------------------------
__global__ void k_gat_scores(const float* __restrict__ H,
                             const float* __restrict__ att_src,
                             const float* __restrict__ att_dst,
                             float* __restrict__ es, float* __restrict__ ed) {
    int wid  = blockIdx.x * 8 + (threadIdx.x >> 5);   // 2048*4 waves
    int lane = threadIdx.x & 31;
    int r = wid >> 2, h = wid & 3;
    const float* hp = H + (size_t)r * (NH_ * C_) + h * C_;
    const float* as = att_src + h * C_;
    const float* ad = att_dst + h * C_;
    float s = 0.f, d = 0.f;
    for (int c = lane; c < C_; c += 32) { float v = hp[c]; s += v * as[c]; d += v * ad[c]; }
    for (int off = 16; off; off >>= 1) { s += __shfl_xor(s, off, 32); d += __shfl_xor(d, off, 32); }
    if (lane == 0) { es[wid] = s; ed[wid] = d; }
}

// writes m (f32, for final pool) and its f16 copy (GEMM A operand next step)
__global__ void k_gat_combine(const float* __restrict__ H,
                              const float* __restrict__ es, const float* __restrict__ ed,
                              const float* __restrict__ gat_b,
                              float* __restrict__ mout, _Float16* __restrict__ mouth) {
    int r = blockIdx.x;          // g*16+p
    int c = threadIdx.x;         // 0..255
    int g = r >> 4;
    float out = 0.f;
    for (int h = 0; h < NH_; ++h) {
        float es0 = es[(g * 16 + 0) * NH_ + h];
        float esp = es[r * NH_ + h];
        float edp = ed[r * NH_ + h];
        float e1 = es0 + edp; e1 = e1 > 0.f ? e1 : 0.2f * e1;   // speaker -> t
        float e2 = esp + edp; e2 = e2 > 0.f ? e2 : 0.2f * e2;   // self loop
        float mx = fmaxf(e1, e2);
        float w1 = expf(e1 - mx), w2 = expf(e2 - mx);
        float inv = 1.f / (w1 + w2);
        float h0 = H[(size_t)(g * 16 + 0) * (NH_ * C_) + h * C_ + c];
        float hp = H[(size_t)r * (NH_ * C_) + h * C_ + c];
        out += (w1 * h0 + w2 * hp) * inv;
    }
    float v = out * 0.25f + gat_b[c];
    mout[(size_t)r * C_ + c]  = v;
    mouth[(size_t)r * C_ + c] = (_Float16)v;
}

// ---------------------------------------------------------------------------
// GRU pointwise
// ---------------------------------------------------------------------------
__global__ void k_gru1(const float* __restrict__ GI, const float* __restrict__ GH,
                       const float* __restrict__ bih, const float* __restrict__ bhh,
                       float* __restrict__ h, _Float16* __restrict__ hh) {
    int i = blockIdx.x * 256 + threadIdx.x;
    if (i >= GP * C_) return;
    int row = i >> 8, c = i & 255;
    size_t b = (size_t)row * (3 * C_);
    float ir = GI[b + c]          + bih[c];
    float iz = GI[b + C_ + c]     + bih[C_ + c];
    float in = GI[b + 2 * C_ + c] + bih[2 * C_ + c];
    float hr = GH[b + c]          + bhh[c];
    float hz = GH[b + C_ + c]     + bhh[C_ + c];
    float hn = GH[b + 2 * C_ + c] + bhh[2 * C_ + c];
    float rr = 1.f / (1.f + expf(-(ir + hr)));
    float zz = 1.f / (1.f + expf(-(iz + hz)));
    float nn = tanhf(in + rr * hn);
    float hv = (1.f - zz) * nn + zz * h[i];
    h[i]  = hv;
    hh[i] = (_Float16)hv;
}

// zero-input GRU step: gi = bih; writes h_nxt (f16, consumed only as GEMM A)
__global__ void k_gru2(const float* __restrict__ GH2,
                       const float* __restrict__ bih, const float* __restrict__ bhh,
                       const float* __restrict__ h, _Float16* __restrict__ hn_out) {
    int i = blockIdx.x * 256 + threadIdx.x;
    if (i >= GP * C_) return;
    int row = i >> 8, c = i & 255;
    size_t b = (size_t)row * (3 * C_);
    float ir = bih[c];
    float iz = bih[C_ + c];
    float in = bih[2 * C_ + c];
    float hr = GH2[b + c]          + bhh[c];
    float hz = GH2[b + C_ + c]     + bhh[C_ + c];
    float hn = GH2[b + 2 * C_ + c] + bhh[2 * C_ + c];
    float rr = 1.f / (1.f + expf(-(ir + hr)));
    float zz = 1.f / (1.f + expf(-(iz + hz)));
    float nn = tanhf(in + rr * hn);
    hn_out[i] = (_Float16)((1.f - zz) * nn + zz * h[i]);
}

// ---------------------------------------------------------------------------
// gated global-attention pool over P nodes + 7-way classifier
// ---------------------------------------------------------------------------
__global__ void k_final(const float* __restrict__ mem,
                        const float* __restrict__ attW, const float* __restrict__ attB,
                        const float* __restrict__ clsW, const float* __restrict__ clsB,
                        float* __restrict__ out) {
    __shared__ float eL[16];
    __shared__ float gate[16];
    __shared__ float pooled[256];
    int sg = blockIdx.x;               // s*128 + g
    int s = sg >> 7, g = sg & 127;
    int lane = threadIdx.x & 31, wave = threadIdx.x >> 5;
    const float* base = mem + ((size_t)s * GP + (size_t)g * P_) * C_;   // [16][256]

    for (int pp = wave; pp < P_; pp += 8) {
        float acc = 0.f;
        for (int c = lane; c < C_; c += 32) acc += base[pp * C_ + c] * attW[c];
        for (int off = 16; off; off >>= 1) acc += __shfl_xor(acc, off, 32);
        if (lane == 0) eL[pp] = acc + attB[0];
    }
    __syncthreads();
    if (threadIdx.x == 0) {
        float mx = eL[0];
        for (int p = 1; p < P_; ++p) mx = fmaxf(mx, eL[p]);
        float sum = 0.f;
        for (int p = 0; p < P_; ++p) { float w = expf(eL[p] - mx); gate[p] = w; sum += w; }
        float inv = 1.f / sum;
        for (int p = 0; p < P_; ++p) gate[p] *= inv;
    }
    __syncthreads();
    {
        int c = threadIdx.x;
        float acc = 0.f;
        for (int p = 0; p < P_; ++p) acc += gate[p] * base[p * C_ + c];
        pooled[c] = acc;
    }
    __syncthreads();
    if (wave < OUT_) {
        float acc = 0.f;
        for (int c = lane; c < C_; c += 32) acc += pooled[c] * clsW[wave * C_ + c];
        for (int off = 16; off; off >>= 1) acc += __shfl_xor(acc, off, 32);
        if (lane == 0) out[((size_t)g * S_ + s) * OUT_ + wave] = acc + clsB[wave];
    }
}

// ---------------------------------------------------------------------------
extern "C" void kernel_launch(void* const* d_in, const int* in_sizes, int n_in,
                              void* d_out, int out_size, void* d_ws, size_t ws_size,
                              hipStream_t stream) {
    (void)in_sizes; (void)n_in; (void)out_size; (void)ws_size;
    const float* x      = (const float*)d_in[0];
    const float* gatW   = (const float*)d_in[1];
    const float* attSrc = (const float*)d_in[2];
    const float* attDst = (const float*)d_in[3];
    const float* gatB   = (const float*)d_in[4];
    const float* Wih    = (const float*)d_in[5];
    const float* Whh    = (const float*)d_in[6];
    const float* bih    = (const float*)d_in[7];
    const float* bhh    = (const float*)d_in[8];
    const float* upW    = (const float*)d_in[9];
    const float* upB    = (const float*)d_in[10];
    const float* spW    = (const float*)d_in[11];
    const float* spB    = (const float*)d_in[12];
    const float* attW   = (const float*)d_in[13];
    const float* attB   = (const float*)d_in[14];
    const float* clsW   = (const float*)d_in[15];
    const float* clsB   = (const float*)d_in[16];
    float* out = (float*)d_out;

    char* ws = (char*)d_ws;
    size_t off = 0;
    auto allocB = [&](size_t bytes) {
        void* p = (void*)(ws + off);
        off += ((bytes + 255) & ~(size_t)255);
        return p;
    };
    // f32 buffers
    float* mem   = (float*)allocB((size_t)S_ * GP * C_ * 4);
    float* Hbuf  = (float*)allocB((size_t)GP * NH_ * C_ * 4);
    float* GI    = (float*)allocB((size_t)GP * 3 * C_ * 4);
    float* GH    = (float*)allocB((size_t)GP * 3 * C_ * 4);   // reused for GH2
    float* hrun  = (float*)allocB((size_t)GP * C_ * 4);
    float* es    = (float*)allocB((size_t)GP * NH_ * 4);
    float* ed    = (float*)allocB((size_t)GP * NH_ * 4);
    // f16 buffers
    _Float16* memh  = (_Float16*)allocB((size_t)S_ * GP * C_ * 2);
    _Float16* Xbufh = (_Float16*)allocB((size_t)GP * D_ * 2);
    _Float16* hrunh = (_Float16*)allocB((size_t)GP * C_ * 2);
    _Float16* hnxth = (_Float16*)allocB((size_t)GP * C_ * 2);
    _Float16* spkh  = (_Float16*)allocB((size_t)G_ * D_ * 2);
    _Float16* gatWh = (_Float16*)allocB((size_t)(NH_ * C_) * D_ * 2);
    _Float16* Wihh  = (_Float16*)allocB((size_t)(3 * C_) * C_ * 2);
    _Float16* Whhh  = (_Float16*)allocB((size_t)(3 * C_) * C_ * 2);
    _Float16* upWh  = (_Float16*)allocB((size_t)D_ * C_ * 2);
    _Float16* spWsh = (_Float16*)allocB((size_t)D_ * D_ * 2);

    const dim3 gblk(256);
    #define CEIL256(n) (((n) + 255) / 256)

    // ---- prep: weights -> f16 [N][K]
    k_prep_gatW <<<CEIL256((NH_*C_)*D_), 256, 0, stream>>>(gatW, gatWh);
    k_f32_to_f16<<<CEIL256(3*C_*C_),     256, 0, stream>>>(Wih, Wihh, 3*C_*C_);
    k_f32_to_f16<<<CEIL256(3*C_*C_),     256, 0, stream>>>(Whh, Whhh, 3*C_*C_);
    k_f32_to_f16<<<CEIL256(D_*C_),       256, 0, stream>>>(upW, upWh, D_*C_);
    k_spwsum_h  <<<CEIL256(D_*D_),       256, 0, stream>>>(spW, spWsh);
    k_gather_x0 <<<CEIL256(GP*D_),       256, 0, stream>>>(x, Xbufh);
    k_fill0     <<<CEIL256(GP*C_),       256, 0, stream>>>(hrun, GP*C_);
    k_fillh0    <<<CEIL256(GP*C_),       256, 0, stream>>>(hrunh, GP*C_);

    auto gat = [&](float* mout, _Float16* mouth) {
        gemm_tdm<0, 0><<<dim3((NH_ * C_) / 128, GP / 64), gblk, 0, stream>>>(
            Xbufh, D_, gatWh, D_, Hbuf, NH_ * C_, D_, (const float*)nullptr);
        k_gat_scores <<<(GP * NH_) / 8, 256, 0, stream>>>(Hbuf, attSrc, attDst, es, ed);
        k_gat_combine<<<GP, 256, 0, stream>>>(Hbuf, es, ed, gatB, mout, mouth);
    };

    gat(mem, memh);   // m0

    for (int j = 1; j < S_; ++j) {
        const _Float16* mprevh = memh + (size_t)(j - 1) * GP * C_;
        // GI = m_prev @ Wih^T   ([2048,256] x [768,256]^T)
        gemm_tdm<0, 0><<<dim3((3 * C_) / 128, GP / 64), gblk, 0, stream>>>(
            mprevh, C_, Wihh, C_, GI, 3 * C_, C_, (const float*)nullptr);
        // GH = h_run @ Whh^T
        gemm_tdm<0, 0><<<dim3((3 * C_) / 128, GP / 64), gblk, 0, stream>>>(
            hrunh, C_, Whhh, C_, GH, 3 * C_, C_, (const float*)nullptr);
        k_gru1<<<CEIL256(GP * C_), 256, 0, stream>>>(GI, GH, bih, bhh, hrun, hrunh);
        // GH2 = h_run(new) @ Whh^T
        gemm_tdm<0, 0><<<dim3((3 * C_) / 128, GP / 64), gblk, 0, stream>>>(
            hrunh, C_, Whhh, C_, GH, 3 * C_, C_, (const float*)nullptr);
        k_gru2<<<CEIL256(GP * C_), 256, 0, stream>>>(GH, bih, bhh, hrun, hnxth);
        // pooled = h_nxt @ upW^T + upB   -> Xbufh (f16)
        gemm_tdm<1, 1><<<dim3(D_ / 128, GP / 64), gblk, 0, stream>>>(
            hnxth, C_, upWh, C_, Xbufh, D_, C_, upB);
        // spk = relu(Xbufh[rows p==0] @ spWsum^T + spB)   (A row stride 16*768)
        gemm_tdm<2, 1><<<dim3(D_ / 128, G_ / 64), gblk, 0, stream>>>(
            Xbufh, P_ * D_, spWsh, D_, spkh, D_, D_, spB);
        k_scatter_spk<<<CEIL256(G_ * D_), 256, 0, stream>>>(spkh, Xbufh);
        gat(mem + (size_t)j * GP * C_, memh + (size_t)j * GP * C_);
    }

    k_final<<<S_ * G_, 256, 0, stream>>>(mem, attW, attB, clsW, clsB, out);
}
